// TinyFormerEncoder_33311766347978
// MI455X (gfx1250) — matrix operand order repaired and checked by reading, not compiled, not run on hardware
//
#include <hip/hip_runtime.h>
#include <hip/hip_bf16.h>

typedef _Float16 half_t;
typedef half_t v16h __attribute__((ext_vector_type(16)));
typedef half_t v8h  __attribute__((ext_vector_type(8)));
typedef float  v8f  __attribute__((ext_vector_type(8)));

#define DEV static __device__ __forceinline__

// Wave-internal LDS fence: LDS ops from one wave complete in order; after
// dscnt==0 all prior LDS writes (any lane) are visible to all lanes of the wave.
DEV void lds_fence() { asm volatile("s_wait_dscnt 0x0" ::: "memory"); }

// A-fragment (16x32 f16, MxK) from row-major [M][rowStride] LDS storage.
// Layout (ISA 7.12.2): lane l: m = l&15, half = l>>4;
// element e: K = kbase + (e<8?0:16) + half*8 + (e&7)  -> two contiguous 8-half chunks.
DEV v16h load_frag_A(const half_t* base, int rowStride, int kbase, int lane) {
  const int hf = lane >> 4, m = lane & 15;
  v8h lo = *(const v8h*)(base + m * rowStride + kbase + hf * 8);
  v8h hi = *(const v8h*)(base + m * rowStride + kbase + 16 + hf * 8);
  return __builtin_shufflevector(lo, hi, 0,1,2,3,4,5,6,7,8,9,10,11,12,13,14,15);
}

// B-fragment (32x16 f16, KxN) from [N][rowStride] ("transposed") LDS storage.
// Layout: lane l: n = l&15, half = l>>4; element e: K = kbase + (e<8?0:16) + half*8 + (e&7)
// -> two contiguous 8-half chunks (matches A).
DEV v16h load_frag_B(const half_t* base, int rowStride, int n0, int kbase, int lane) {
  const int hf = lane >> 4, n = lane & 15;
  const half_t* p = base + (n0 + n) * rowStride + kbase + hf * 8;
  v8h lo = *(const v8h*)(p);
  v8h hi = *(const v8h*)(p + 16);
  return __builtin_shufflevector(lo, hi, 0,1,2,3,4,5,6,7,8,9,10,11,12,13,14,15);
}

DEV v8f splat8(float x) {
  v8f c;
#pragma unroll
  for (int r = 0; r < 8; ++r) c[r] = x;
  return c;
}

DEV v8f wmma_f16(v16h a, v16h b, v8f c) {
  return __builtin_amdgcn_wmma_f32_16x16x32_f16(false, a, false, b, (short)0, c, false, false);
}

// C/D layout: VGPR r holds element (m = half*8 + r, n = lane&15).
DEV void store_d_rm_f16(half_t* base, int rowStride, int n0, v8f d, int lane) {
  const int hf = lane >> 4, n = lane & 15;
#pragma unroll
  for (int r = 0; r < 8; ++r) base[(hf * 8 + r) * rowStride + n0 + n] = (half_t)d[r];
}

// Store C/D tile transposed: dst[n][m] with row stride mStride; per-lane contiguous.
DEV void store_d_tr_f16(half_t* base, int mStride, int n0, v8f d, int lane) {
  const int hf = lane >> 4, n = lane & 15;
  v8h t;
#pragma unroll
  for (int r = 0; r < 8; ++r) t[r] = (half_t)d[r];
  *(v8h*)(base + (n0 + n) * mStride + hf * 8) = t;
}

struct alignas(16) Weights {
  half_t WqT[1024], WkT[1024], WvT[1024], WoT[1024]; // [32 out][32 in]
  half_t W1T[2048];                                  // [64 out][32 in]
  half_t W2T[2048];                                  // [32 out][64 in]
  float  bias[224];  // bq@0 bk@32 bv@64 bo@96 b1@128 b2@192
};

struct alignas(16) Scratch {
  half_t a[512];   // x(f16)  -> attn (16x32, cols 16..31 zero)
  half_t b[512];   // q       -> context     (row-major 16x32)
  half_t c[512];   // k       -> y(f16)      (row-major 16x32)
  half_t d[1024];  // v^T (32x32, t-padded) -> h (16x64)
  float  xf[512];  // x f32 (residual 1)
  float  yf[512];  // y f32 (residual 2)
  float  sc[256];  // scores f32 16x16
};

__global__ __launch_bounds__(128) void tinyformer_encoder_kernel(
    const float* __restrict__ x,
    const float* __restrict__ wq, const float* __restrict__ bq,
    const float* __restrict__ wk, const float* __restrict__ bk,
    const float* __restrict__ wv, const float* __restrict__ bv,
    const float* __restrict__ wo, const float* __restrict__ bo,
    const float* __restrict__ w1, const float* __restrict__ b1,
    const float* __restrict__ w2, const float* __restrict__ b2,
    float* __restrict__ out, int nB)
{
  __shared__ Weights sW;
  __shared__ Scratch sS[4];

  const int tid = threadIdx.x;

  // ---- Stage weights into LDS as f16, transposed to [out][in] ----
  for (int i = tid; i < 1024; i += 128) {
    int o = i >> 5, in = i & 31;                    // weights stored [in,out] in memory
    sW.WqT[i] = (half_t)wq[in * 32 + o];
    sW.WkT[i] = (half_t)wk[in * 32 + o];
    sW.WvT[i] = (half_t)wv[in * 32 + o];
    sW.WoT[i] = (half_t)wo[in * 32 + o];
  }
  for (int i = tid; i < 2048; i += 128) {
    int o1 = i >> 5, in1 = i & 31;                  // W1T[o<64][in<32]
    sW.W1T[i] = (half_t)w1[in1 * 64 + o1];
    int o2 = i >> 6, in2 = i & 63;                  // W2T[o<32][in<64]
    sW.W2T[i] = (half_t)w2[in2 * 32 + o2];
  }
  for (int i = tid; i < 224; i += 128) {
    float v;
    if      (i < 32)  v = bq[i];
    else if (i < 64)  v = bk[i - 32];
    else if (i < 96)  v = bv[i - 64];
    else if (i < 128) v = bo[i - 96];
    else if (i < 192) v = b1[i - 128];
    else              v = b2[i - 192];
    sW.bias[i] = v;
  }
  __syncthreads();

  const int lane = tid & 31;
  const int wave = tid >> 5;
  const int hf = lane >> 4;
  const int n  = lane & 15;
  Scratch& S = sS[wave];

  const float rscale = 0.17677669529663687f;  // 1/sqrt(D), D=32
  const int wstride = gridDim.x * 4;

  for (int b = blockIdx.x * 4 + wave; b < nB; b += wstride) {
    const float* xb = x + (size_t)b * 512;
    // Prefetch next iteration's x tile (global_prefetch_b8).
    if (b + wstride < nB)
      __builtin_prefetch(x + (size_t)(b + wstride) * 512 + lane * 16, 0, 0);

    // ---- 1. Load x (16x32 f32): 16 consecutive floats per lane ----
    const float4* xp = (const float4*)(xb + lane * 16);
    float4 x0 = xp[0], x1 = xp[1], x2 = xp[2], x3 = xp[3];
    float4* fp = (float4*)(S.xf + lane * 16);
    fp[0] = x0; fp[1] = x1; fp[2] = x2; fp[3] = x3;
    v8h e0, e1;
    e0[0] = (half_t)x0.x; e0[1] = (half_t)x0.y; e0[2] = (half_t)x0.z; e0[3] = (half_t)x0.w;
    e0[4] = (half_t)x1.x; e0[5] = (half_t)x1.y; e0[6] = (half_t)x1.z; e0[7] = (half_t)x1.w;
    e1[0] = (half_t)x2.x; e1[1] = (half_t)x2.y; e1[2] = (half_t)x2.z; e1[3] = (half_t)x2.w;
    e1[4] = (half_t)x3.x; e1[5] = (half_t)x3.y; e1[6] = (half_t)x3.z; e1[7] = (half_t)x3.w;
    *(v8h*)(S.a + lane * 16)     = e0;
    *(v8h*)(S.a + lane * 16 + 8) = e1;
    // Zero-pad v^T columns t=16..31 (lane zeroes its own row of S.d).
    v8h z8;
#pragma unroll
    for (int i = 0; i < 8; ++i) z8[i] = (half_t)0.f;
    *(v8h*)(S.d + lane * 32 + 16) = z8;
    *(v8h*)(S.d + lane * 32 + 24) = z8;
    lds_fence();

    // ---- 2. Q/K/V projections: q = x@Wq + bq, etc. (6 WMMAs) ----
    v16h aX = load_frag_A(S.a, 32, 0, lane);
    v8f dq0 = wmma_f16(aX, load_frag_B(sW.WqT, 32, 0,  0, lane), splat8(sW.bias[0  + n]));
    v8f dq1 = wmma_f16(aX, load_frag_B(sW.WqT, 32, 16, 0, lane), splat8(sW.bias[16 + n]));
    store_d_rm_f16(S.b, 32, 0, dq0, lane);
    store_d_rm_f16(S.b, 32, 16, dq1, lane);
    v8f dk0 = wmma_f16(aX, load_frag_B(sW.WkT, 32, 0,  0, lane), splat8(sW.bias[32 + n]));
    v8f dk1 = wmma_f16(aX, load_frag_B(sW.WkT, 32, 16, 0, lane), splat8(sW.bias[48 + n]));
    store_d_rm_f16(S.c, 32, 0, dk0, lane);
    store_d_rm_f16(S.c, 32, 16, dk1, lane);
    v8f dv0 = wmma_f16(aX, load_frag_B(sW.WvT, 32, 0,  0, lane), splat8(sW.bias[64 + n]));
    v8f dv1 = wmma_f16(aX, load_frag_B(sW.WvT, 32, 16, 0, lane), splat8(sW.bias[80 + n]));
    store_d_tr_f16(S.d, 32, 0,  dv0, lane);  // v^T[d][t]
    store_d_tr_f16(S.d, 32, 16, dv1, lane);
    lds_fence();

    // ---- 3. scores = q @ k^T (1 WMMA); B[K=d][N=t] = k row-major ----
    v16h aQ = load_frag_A(S.b, 32, 0, lane);
    v16h bK = load_frag_B(S.c, 32, 0, 0, lane);
    v8f sc8 = wmma_f16(aQ, bK, splat8(0.f));
#pragma unroll
    for (int r = 0; r < 8; ++r) S.sc[(hf * 8 + r) * 16 + n] = sc8[r];
    lds_fence();

    // ---- 4. Softmax over rows (lanes 0..15, one row each) ----
    if (lane < 16) {
      float v[16];
      float mx = -3.4e38f;
#pragma unroll
      for (int j = 0; j < 16; ++j) { v[j] = S.sc[lane * 16 + j] * rscale; mx = fmaxf(mx, v[j]); }
      float sum = 0.f;
#pragma unroll
      for (int j = 0; j < 16; ++j) { float e = __expf(v[j] - mx); v[j] = e; sum += e; }
      float inv = 1.f / sum;
      half_t* ar = S.a + lane * 32;
#pragma unroll
      for (int j = 0; j < 16; ++j) ar[j] = (half_t)(v[j] * inv);
      *(v8h*)(ar + 16) = z8;  // zero-pad K=16..31
      *(v8h*)(ar + 24) = z8;
    }
    lds_fence();

    // ---- 5. context = attn @ v (2 WMMAs); B from v^T, K padded to 32 ----
    v16h aP = load_frag_A(S.a, 32, 0, lane);
    v8f c0 = wmma_f16(aP, load_frag_B(S.d, 32, 0,  0, lane), splat8(0.f));
    v8f c1 = wmma_f16(aP, load_frag_B(S.d, 32, 16, 0, lane), splat8(0.f));
    store_d_rm_f16(S.b, 32, 0, c0, lane);
    store_d_rm_f16(S.b, 32, 16, c1, lane);
    lds_fence();

    // ---- 6. y = x + context@Wo + bo (2 WMMAs + residual) ----
    v16h aC = load_frag_A(S.b, 32, 0, lane);
#pragma unroll
    for (int t2 = 0; t2 < 2; ++t2) {
      int n0 = t2 * 16;
      v8f acc = splat8(sW.bias[96 + n0 + n]);
      acc = wmma_f16(aC, load_frag_B(sW.WoT, 32, n0, 0, lane), acc);
#pragma unroll
      for (int r = 0; r < 8; ++r) {
        int idx = (hf * 8 + r) * 32 + n0 + n;
        float yv = S.xf[idx] + acc[r];
        S.yf[idx] = yv;
        S.c[idx]  = (half_t)yv;
      }
    }
    lds_fence();

    // ---- 7. h = relu(y@W1 + b1) (4 WMMAs, N=64) ----
    v16h aY = load_frag_A(S.c, 32, 0, lane);
#pragma unroll
    for (int t4 = 0; t4 < 4; ++t4) {
      int n0 = t4 * 16;
      v8f acc = splat8(sW.bias[128 + n0 + n]);
      acc = wmma_f16(aY, load_frag_B(sW.W1T, 32, n0, 0, lane), acc);
#pragma unroll
      for (int r = 0; r < 8; ++r)
        S.d[(hf * 8 + r) * 64 + n0 + n] = (half_t)fmaxf(acc[r], 0.f);
    }
    lds_fence();

    // ---- 8. z = y + h@W2 + b2 (K=64 -> 2 accumulating WMMAs x 2 tiles) ----
    v16h aH0 = load_frag_A(S.d, 64, 0,  lane);
    v16h aH1 = load_frag_A(S.d, 64, 32, lane);
    float* ob = out + (size_t)b * 512;
#pragma unroll
    for (int t2 = 0; t2 < 2; ++t2) {
      int n0 = t2 * 16;
      v8f acc = splat8(sW.bias[192 + n0 + n]);
      acc = wmma_f16(aH0, load_frag_B(sW.W2T, 64, n0, 0,  lane), acc);
      acc = wmma_f16(aH1, load_frag_B(sW.W2T, 64, n0, 32, lane), acc);
#pragma unroll
      for (int r = 0; r < 8; ++r) {
        int idx = (hf * 8 + r) * 32 + n0 + n;
        ob[idx] = S.yf[idx] + acc[r];
      }
    }
  }
}

extern "C" void kernel_launch(void* const* d_in, const int* in_sizes, int n_in,
                              void* d_out, int out_size, void* d_ws, size_t ws_size,
                              hipStream_t stream) {
  const float* x  = (const float*)d_in[0];
  const float* wq = (const float*)d_in[1];
  const float* bq = (const float*)d_in[2];
  const float* wk = (const float*)d_in[3];
  const float* bk = (const float*)d_in[4];
  const float* wv = (const float*)d_in[5];
  const float* bv = (const float*)d_in[6];
  const float* wo = (const float*)d_in[7];
  const float* bo = (const float*)d_in[8];
  const float* w1 = (const float*)d_in[9];
  const float* b1 = (const float*)d_in[10];
  const float* w2 = (const float*)d_in[11];
  const float* b2 = (const float*)d_in[12];
  float* out = (float*)d_out;

  int nB = in_sizes[0] / (16 * 32);   // B = 65536
  int blocks = (nB + 3) / 4;          // 4 waves (batch elems) per 128-thread block
  if (blocks > 4096) blocks = 4096;   // grid-stride; amortizes weight staging
  if (blocks < 1) blocks = 1;

  tinyformer_encoder_kernel<<<blocks, 128, 0, stream>>>(
      x, wq, bq, wk, bk, wv, bv, wo, bo, w1, b1, w2, b2, out, nB);
}
